// VectorQuantizer_88553635709134
// MI455X (gfx1250) — compile-verified, matrix operand order checked by reading, not run on hardware
//
#include <hip/hip_runtime.h>
#include <hip/hip_bf16.h>

typedef __attribute__((ext_vector_type(16))) _Float16 v16h;
typedef __attribute__((ext_vector_type(8)))  _Float16 v8h;
typedef __attribute__((ext_vector_type(2)))  _Float16 v2h;
typedef __attribute__((ext_vector_type(8)))  float    v8f;

#define VQ_B   32
#define VQ_C   64
#define VQ_HW  4096             // H*W
#define VQ_N   (VQ_B * VQ_HW)   // 131072 flat rows
#define VQ_K   512              // codebook entries
#define VQ_OUTQ (VQ_N * VQ_C)   // 8388608 quantized elements

#define LDS_ROW 72              // half stride per row: 144B -> 16B aligned, spreads banks
#define ROWS_PER_BLOCK 256      // 8 waves x 2 M-tiles x 16 rows

// Build a v16h in the 16-bit A/B WMMA lane layout from a 64-half row:
// lanes 0-15 take K={off..off+7, off+16..off+23}, lanes 16-31 take K={off+8.., off+24..}
__device__ __forceinline__ v16h load_tile_row(const _Float16* __restrict__ row,
                                              int hi, int off) {
    v8h lo = *(const v8h*)(row + off +      hi * 8);
    v8h hh = *(const v8h*)(row + off + 16 + hi * 8);
    return __builtin_shufflevector(lo, hh, 0,1,2,3,4,5,6,7,8,9,10,11,12,13,14,15);
}

// ---------------------------------------------------------------------------
// Kernel 1: zero loss accumulator + L2-normalize codebook into f16 (one wave/row)
// ---------------------------------------------------------------------------
__global__ __launch_bounds__(128) void vq_norm_init(const float* __restrict__ emb,
                                                    _Float16* __restrict__ embN,
                                                    float* __restrict__ acc) {
    if (blockIdx.x == 0 && threadIdx.x == 0) acc[0] = 0.0f;
    const int wave = threadIdx.x >> 5;
    const int lane = threadIdx.x & 31;
    const int row  = blockIdx.x * 4 + wave;           // 128 blocks * 4 waves = 512 rows
    const float* r = emb + row * VQ_C;
    float a = r[lane];
    float b = r[lane + 32];
    float s = a * a + b * b;
    #pragma unroll
    for (int off = 16; off; off >>= 1) s += __shfl_xor(s, off, 32);
    float inv = 1.0f / fmaxf(sqrtf(s), 1e-12f);
    embN[row * VQ_C + lane]      = (_Float16)(a * inv);
    embN[row * VQ_C + lane + 32] = (_Float16)(b * inv);
}

// ---------------------------------------------------------------------------
// Kernel 2: WMMA argmax.  Block = 256 threads (8 waves), each wave owns two
// 16-row M-tiles (256 rows/block).  B tiles double-buffered in registers so
// global loads for tile t+1 overlap the 4 WMMAs + argmax VALU of tile t.
// Argmax update: one v_cmp + two v_cndmask per element (no fmax/NaN
// canonicalization) — the minimum for value+index tracking.
// ---------------------------------------------------------------------------
__global__ __launch_bounds__(256) void vq_argmax(const float* __restrict__ z,
                                                 const _Float16* __restrict__ embN,
                                                 int* __restrict__ idx_ws,
                                                 float* __restrict__ out_idx) {
    __shared__ alignas(16) _Float16 lds[ROWS_PER_BLOCK * LDS_ROW];

    const int tid   = threadIdx.x;
    const int nbase = blockIdx.x * ROWS_PER_BLOCK;
    const int b     = nbase >> 12;            // / HW
    const int hw0   = nbase & (VQ_HW - 1);
    const float* zb = z + ((size_t)b << 18);  // b * C * HW

    // Stage + transpose z tile into LDS as f16 rows [local_n][c]; packed half2
    // stores: each thread handles a (row, channel-pair) per iteration.
    #pragma unroll 4
    for (int it = 0; it < 32; ++it) {
        int p  = it * 256 + tid;     // pair index < 8192
        int cp = p >> 8;             // channel pair 0..31
        int r  = p & 255;            // local row
        float v0 = zb[((size_t)(2 * cp)     << 12) + hw0 + r];   // coalesced along r
        float v1 = zb[((size_t)(2 * cp + 1) << 12) + hw0 + r];
        v2h pair = { (_Float16)v0, (_Float16)v1 };
        *(v2h*)(&lds[r * LDS_ROW + 2 * cp]) = pair;              // ds_store_b32
    }
    __syncthreads();

    const int wave = tid >> 5;
    const int lane = tid & 31;
    const int m    = lane & 15;
    const int hi   = lane >> 4;

    // Two A tiles per wave, kept in registers
    v16h a1[2], a2[2];
    #pragma unroll
    for (int mt = 0; mt < 2; ++mt) {
        const _Float16* arow = &lds[(wave * 32 + mt * 16 + m) * LDS_ROW];
        a1[mt] = load_tile_row(arow, hi, 0);
        a2[mt] = load_tile_row(arow, hi, 32);
    }

    v8f best[2];
    int bidx[2][8];
    #pragma unroll
    for (int mt = 0; mt < 2; ++mt)
        #pragma unroll
        for (int v = 0; v < 8; ++v) { best[mt][v] = -3.402823466e38f; bidx[mt][v] = 0; }

    // Prologue: load B tile 0
    const _Float16* brow0 = embN + m * VQ_C;
    v16h b1c = load_tile_row(brow0, hi, 0);
    v16h b2c = load_tile_row(brow0, hi, 32);

    #pragma unroll 2
    for (int t = 0; t < 32; ++t) {
        // Issue loads for next tile (wraps to 0 on last iter; harmless)
        const _Float16* brow = embN + (((t + 1) & 31) * 16 + m) * VQ_C;
        v16h b1n = load_tile_row(brow, hi, 0);
        v16h b2n = load_tile_row(brow, hi, 32);

        const int col = t * 16 + m;           // C/D layout: N = lane & 15
        #pragma unroll
        for (int mt = 0; mt < 2; ++mt) {
            v8f accv = {};
            accv = __builtin_amdgcn_wmma_f32_16x16x32_f16(false, a1[mt], false, b1c,
                                                          (short)0, accv, false, false);
            accv = __builtin_amdgcn_wmma_f32_16x16x32_f16(false, a2[mt], false, b2c,
                                                          (short)0, accv, false, false);
            #pragma unroll
            for (int v = 0; v < 8; ++v) {
                bool gt = accv[v] > best[mt][v];
                best[mt][v] = gt ? accv[v] : best[mt][v];
                bidx[mt][v] = gt ? col : bidx[mt][v];
            }
        }
        b1c = b1n; b2c = b2n;
    }

    // Reduce (max, argmax) across the 16 lanes of each column group (branchless)
    #pragma unroll
    for (int off = 8; off; off >>= 1) {
        #pragma unroll
        for (int mt = 0; mt < 2; ++mt) {
            #pragma unroll
            for (int v = 0; v < 8; ++v) {
                float ov = __shfl_xor(best[mt][v], off, 32);
                int   oi = __shfl_xor(bidx[mt][v], off, 32);
                bool take = (ov > best[mt][v]) | ((ov == best[mt][v]) & (oi < bidx[mt][v]));
                best[mt][v] = take ? ov : best[mt][v];
                bidx[mt][v] = take ? oi : bidx[mt][v];
            }
        }
    }

    // Lane 0 holds rows M=0..7, lane 16 holds rows M=8..15 of each tile
    if (m == 0) {
        #pragma unroll
        for (int mt = 0; mt < 2; ++mt) {
            #pragma unroll
            for (int v = 0; v < 8; ++v) {
                int n = nbase + wave * 32 + mt * 16 + hi * 8 + v;
                idx_ws[n]  = bidx[mt][v];
                out_idx[n] = (float)bidx[mt][v];
            }
        }
    }
}

// ---------------------------------------------------------------------------
// Kernel 3: gather + transpose store + MSE accumulation (float4 per thread)
// out[b][c][hw] = emb[idx[b*HW+hw]][c]; coalesced along hw.
// ---------------------------------------------------------------------------
__global__ __launch_bounds__(256) void vq_gather(const float* __restrict__ z,
                                                 const float* __restrict__ emb,
                                                 const int* __restrict__ idx_ws,
                                                 float* __restrict__ out_q,
                                                 float* __restrict__ acc) {
    const int o  = (blockIdx.x * 256 + threadIdx.x) * 4;  // < 8388608
    const int hw = o & (VQ_HW - 1);
    const int c  = (o >> 12) & 63;
    const int b  = o >> 18;
    const int n  = (b << 12) | hw;
    const int4 k4 = *(const int4*)(idx_ws + n);
    float4 q;
    q.x = emb[k4.x * VQ_C + c];
    q.y = emb[k4.y * VQ_C + c];
    q.z = emb[k4.z * VQ_C + c];
    q.w = emb[k4.w * VQ_C + c];
    *(float4*)(out_q + o) = q;
    const float4 zv = *(const float4*)(z + o);
    float dx = q.x - zv.x, dy = q.y - zv.y, dz = q.z - zv.z, dw = q.w - zv.w;
    float s = dx * dx + dy * dy + dz * dz + dw * dw;
    #pragma unroll
    for (int off = 16; off; off >>= 1) s += __shfl_xor(s, off, 32);
    __shared__ float wsum[8];
    const int wave = threadIdx.x >> 5;
    if ((threadIdx.x & 31) == 0) wsum[wave] = s;
    __syncthreads();
    if (threadIdx.x == 0) {
        float t = 0.0f;
        #pragma unroll
        for (int i = 0; i < 8; ++i) t += wsum[i];
        atomicAdd(acc, t);
    }
}

// ---------------------------------------------------------------------------
// Kernel 4: finalize loss = clip(1.25 * mse, 0, 5)
// ---------------------------------------------------------------------------
__global__ void vq_loss(const float* __restrict__ acc, float* __restrict__ out_loss) {
    float mse = acc[0] / (float)VQ_OUTQ;
    float l = 1.25f * mse;
    out_loss[0] = fminf(fmaxf(l, 0.0f), 5.0f);
}

extern "C" void kernel_launch(void* const* d_in, const int* in_sizes, int n_in,
                              void* d_out, int out_size, void* d_ws, size_t ws_size,
                              hipStream_t stream) {
    (void)in_sizes; (void)n_in; (void)out_size; (void)ws_size;
    const float* z   = (const float*)d_in[0];      // (32,64,64,64)
    const float* emb = (const float*)d_in[1];      // (512,64)

    // workspace layout
    char* ws = (char*)d_ws;
    float*     acc    = (float*)ws;                     // 4 B
    _Float16*  embN   = (_Float16*)(ws + 256);          // 512*64*2 = 64 KB
    int*       idx_ws = (int*)(ws + 256 + 65536);       // 131072*4 = 512 KB

    // output layout (return order: quantized_st, indices, loss)
    float* out_q    = (float*)d_out;                    // 8388608
    float* out_idx  = out_q + VQ_OUTQ;                  // 131072
    float* out_loss = out_idx + VQ_N;                   // 1

    vq_norm_init<<<128, 128, 0, stream>>>(emb, embN, acc);
    vq_argmax  <<<VQ_N / ROWS_PER_BLOCK, 256, 0, stream>>>(z, embN, idx_ws, out_idx);
    vq_gather  <<<VQ_OUTQ / (256 * 4), 256, 0, stream>>>(z, emb, idx_ws, out_q, acc);
    vq_loss    <<<1, 1, 0, stream>>>(acc, out_loss);
}
